// Net_77137612636470
// MI455X (gfx1250) — 8-bit matrix operand order repaired and checked by reading, compiled, not run on hardware
//
#include <hip/hip_runtime.h>
#include <cstdint>
#include <cstddef>

typedef int v8i __attribute__((ext_vector_type(8)));

#define EPSV 1e-5f

__global__ __launch_bounds__(256) void conv0_bn_kernel(
    const float* __restrict__ x, const float* __restrict__ w, const float* __restrict__ bias,
    const float* __restrict__ gamma, const float* __restrict__ beta,
    const float* __restrict__ mean, const float* __restrict__ var,
    float* __restrict__ out)
{
    const int OH = 54, OW = 54, B = 64, CIN = 3, K = 11, S = 4, IN = 224, COUT = 96;
    int idx = blockIdx.x * blockDim.x + threadIdx.x;
    int total = COUT * B * OH * OW;
    if (idx >= total) return;
    int ow = idx % OW; int t = idx / OW;
    int oh = t % OH;  t /= OH;
    int b  = t % B;   int c = t / B;
    float s = bias[c];
    for (int ci = 0; ci < CIN; ++ci) {
        const float* xp = x + ((size_t)(b * CIN + ci) * IN + oh * S) * IN + ow * S;
        const float* wp = w + ((size_t)c * CIN + ci) * (K * K);
        for (int kh = 0; kh < K; ++kh)
            for (int kw = 0; kw < K; ++kw)
                s = fmaf(xp[kh * IN + kw], wp[kh * K + kw], s);
    }
    float v = gamma[c] * (s - mean[c]) * rsqrtf(var[c] + EPSV) + beta[c];
    out[idx] = fminf(fmaxf(v, -1.3f), 1.3f);
}

__global__ __launch_bounds__(256) void pack_w_kernel(
    const float* __restrict__ w, int8_t* __restrict__ wp,
    int Mtiles, int Ktiles, int Kreal)
{
    int t = blockIdx.x * blockDim.x + threadIdx.x;
    int total = Ktiles * Mtiles * 256;
    if (t >= total) return;
    int j    = t & 7;
    int lane = (t >> 3) & 31;
    int tile = t >> 8;
    int kt = tile / Mtiles;
    int mt = tile - kt * Mtiles;
    int m  = mt * 16 + (lane & 15);
    int hi = lane >> 4;
    int kbase = kt * 64 + (j >> 2) * 32 + hi * 16 + (j & 3) * 4;
    uint32_t dw = 0;
    for (int bb = 0; bb < 4; ++bb) {
        int K = kbase + bb;
        int8_t sv = 0;
        if (K < Kreal) {
            float f = w[(size_t)m * Kreal + K];
            sv = (f > 0.f) ? 1 : ((f < 0.f) ? -1 : 0);
        }
        dw |= ((uint32_t)(uint8_t)sv) << (bb * 8);
    }
    ((uint32_t*)wp)[(size_t)tile * 256 + lane * 8 + j] = dw;
}

__global__ __launch_bounds__(256) void pack_act_kernel(
    const float* __restrict__ act, int8_t* __restrict__ outp,
    int Hin, int Win, int OH, int OW, int k, int pad, int Kreal, int Ntiles)
{
    long long t = (long long)blockIdx.x * blockDim.x + threadIdx.x;
    int j    = (int)(t & 7);
    int lane = (int)((t >> 3) & 31);
    long long tile = t >> 8;
    int nt = (int)(tile % Ntiles);
    int kt = (int)(tile / Ntiles);
    int col = lane & 15, half = lane >> 4;
    int n  = nt * 16 + col;
    int hw = OH * OW;
    int b  = n / hw; int rem = n - b * hw;
    int oh = rem / OW; int ow = rem - oh * OW;
    int kk = k * k;
    int kbase = kt * 64 + (j >> 2) * 32 + half * 16 + (j & 3) * 4;
    uint32_t dw = 0;
    for (int bb = 0; bb < 4; ++bb) {
        int K = kbase + bb;
        int8_t sv = 0;
        if (K < Kreal) {
            int ci = K / kk; int r2 = K - ci * kk;
            int kh = r2 / k; int kw = r2 - kh * k;
            int ih = oh - pad + kh, iw = ow - pad + kw;
            if (ih >= 0 && ih < Hin && iw >= 0 && iw < Win) {
                float f = act[(((size_t)ci * 64 + b) * Hin + ih) * Win + iw];
                sv = (f > 0.f) ? 1 : ((f < 0.f) ? -1 : 0);
            }
        }
        dw |= ((uint32_t)(uint8_t)sv) << (bb * 8);
    }
    ((uint32_t*)outp)[(size_t)tile * 256 + lane * 8 + j] = dw;
}

__global__ __launch_bounds__(256) void gemm_bin_wmma(
    const int8_t* __restrict__ ap, const int8_t* __restrict__ bp,
    const float* __restrict__ bias, float* __restrict__ out,
    int Mtiles, int Ntiles, int Ktiles, long long Ntot)
{
    int lane = threadIdx.x & 31;
    int wave = threadIdx.x >> 5;
    const int Mgroups = Mtiles >> 2;
    long long tile  = (long long)blockIdx.x * 8 + wave;
    long long tiles = (long long)Mgroups * Ntiles;
    if (tile >= tiles) return;
    int mg = (int)(tile / Ntiles);
    int nt = (int)(tile - (long long)mg * Ntiles);
    int mt0 = mg << 2;

    const v8i* av = (const v8i*)ap;
    const v8i* bv = (const v8i*)bp;
    v8i acc0 = {0,0,0,0,0,0,0,0};
    v8i acc1 = {0,0,0,0,0,0,0,0};
    v8i acc2 = {0,0,0,0,0,0,0,0};
    v8i acc3 = {0,0,0,0,0,0,0,0};

    for (int kt = 0; kt < Ktiles; ++kt) {
        v8i b = bv[(size_t)((size_t)kt * Ntiles + nt) * 32 + lane];
        if (kt + 1 < Ktiles)
            __builtin_prefetch((const void*)&bv[(size_t)((size_t)(kt + 1) * Ntiles + nt) * 32 + lane], 0, 1);
        size_t abase = (size_t)((size_t)kt * Mtiles + mt0) * 32 + lane;
        v8i a0 = av[abase];
        v8i a1 = av[abase + 32];
        v8i a2 = av[abase + 64];
        v8i a3 = av[abase + 96];
        acc0 = __builtin_amdgcn_wmma_i32_16x16x64_iu8(true, a0, true, b, acc0, false, false);
        acc1 = __builtin_amdgcn_wmma_i32_16x16x64_iu8(true, a1, true, b, acc1, false, false);
        acc2 = __builtin_amdgcn_wmma_i32_16x16x64_iu8(true, a2, true, b, acc2, false, false);
        acc3 = __builtin_amdgcn_wmma_i32_16x16x64_iu8(true, a3, true, b, acc3, false, false);
    }

    int row0 = (lane >> 4) * 8;
    int col  = lane & 15;
    long long ncol = (long long)nt * 16 + col;
    v8i accs[4] = {acc0, acc1, acc2, acc3};
    for (int u = 0; u < 4; ++u) {
        for (int r = 0; r < 8; ++r) {
            int m = (mt0 + u) * 16 + r + row0;
            out[(long long)m * Ntot + ncol] = (float)accs[u][r] + bias[m];
        }
    }
}

__global__ __launch_bounds__(256) void pool_bn_clip_kernel(
    const float* __restrict__ in, float* __restrict__ out,
    const float* __restrict__ g, const float* __restrict__ be,
    const float* __restrict__ mn, const float* __restrict__ vr,
    int C, int H, int W, int pool)
{
    const int B = 64;
    int OH = pool ? H / 2 : H, OW = pool ? W / 2 : W;
    long long total = (long long)C * B * OH * OW;
    long long idx = (long long)blockIdx.x * blockDim.x + threadIdx.x;
    if (idx >= total) return;
    int ow = (int)(idx % OW); long long t = idx / OW;
    int oh = (int)(t % OH);  t /= OH;
    int b  = (int)(t % B);   int c = (int)(t / B);
    float v;
    if (pool) {
        const float* p = in + (((size_t)c * B + b) * H + oh * 2) * W + ow * 2;
        v = fmaxf(fmaxf(p[0], p[1]), fmaxf(p[W], p[W + 1]));
    } else {
        v = in[(((size_t)c * B + b) * H + oh) * W + ow];
    }
    v = g[c] * (v - mn[c]) * rsqrtf(vr[c] + EPSV) + be[c];
    out[idx] = fminf(fmaxf(v, -1.3f), 1.3f);
}

__global__ __launch_bounds__(256) void head_kernel(
    const float* __restrict__ x,
    const float* __restrict__ w,
    const float* __restrict__ bias,
    float* __restrict__ out)
{
    int idx = blockIdx.x * blockDim.x + threadIdx.x;
    if (idx >= 64 * 1000) return;
    int oc = idx % 1000, b = idx / 1000;
    float sm[36];
    float bv = bias[oc];
    for (int p = 0; p < 36; ++p) sm[p] = bv;
    for (int c = 0; c < 1024; ++c) {
        float wv = w[(size_t)oc * 1024 + c];
        const float* xp = x + ((size_t)c * 64 + b) * 36;
        for (int p = 0; p < 36; ++p) sm[p] = fmaf(xp[p], wv, sm[p]);
    }
    float acc = 0.f;
    for (int p = 0; p < 36; ++p) acc += fmaxf(sm[p], 0.f);
    out[(size_t)b * 1000 + oc] = acc * (1.f / 36.f);
}

extern "C" void kernel_launch(void* const* d_in, const int* in_sizes, int n_in,
                              void* d_out, int out_size, void* d_ws, size_t ws_size,
                              hipStream_t stream)
{
    (void)in_sizes; (void)n_in; (void)out_size; (void)ws_size;
    const float* x = (const float*)d_in[0];
    auto P = [&](int layer, int which) -> const float* {
        return (const float*)d_in[1 + layer * 6 + which];
    };
    const float* head_w = (const float*)d_in[1 + 11 * 6];
    const float* head_b = (const float*)d_in[1 + 11 * 6 + 1];

    char* ws = (char*)d_ws;
    const size_t BUF = (size_t)100 << 20;
    const size_t IMC = (size_t)220 << 20;
    float*  bufA   = (float*)(ws);
    float*  bufB   = (float*)(ws + BUF);
    float*  convb  = (float*)(ws + 2 * BUF);
    int8_t* im2col = (int8_t*)(ws + 3 * BUF);
    int8_t* wpack  = (int8_t*)(ws + 3 * BUF + IMC);

    {
        int total = 96 * 64 * 54 * 54;
        conv0_bn_kernel<<<(total + 255) / 256, 256, 0, stream>>>(
            x, P(0, 0), P(0, 1), P(0, 2), P(0, 3), P(0, 4), P(0, 5), bufA);
    }

    int H = 54, W = 54;
    float* cur = bufA;
    float* nxt = bufB;
    struct LCfg { int cin, cout, k, p, pool; };
    const LCfg L[10] = {
        { 96, 128, 3, 1, 0}, {128, 128, 3, 1, 1}, {128, 256, 5, 2, 0},
        {256, 256, 3, 1, 0}, {256, 256, 3, 1, 1}, {256, 384, 3, 1, 0},
        {384, 384, 1, 0, 0}, {384, 384, 1, 0, 1}, {384, 1024, 3, 1, 0},
        {1024, 1024, 1, 0, 0}
    };

    for (int i = 0; i < 10; ++i) {
        const int li = i + 1;
        const int cin = L[i].cin, cout = L[i].cout, k = L[i].k, p = L[i].p, pool = L[i].pool;
        const int Kreal  = cin * k * k;
        const int Ktiles = (Kreal + 63) / 64;
        const int Mtiles = cout / 16;
        const long long N = (long long)64 * H * W;
        const int Ntiles = (int)(N / 16);

        pack_w_kernel<<<(unsigned)(Ktiles * Mtiles), 256, 0, stream>>>(
            P(li, 0), wpack, Mtiles, Ktiles, Kreal);

        pack_act_kernel<<<(unsigned)((long long)Ktiles * Ntiles), 256, 0, stream>>>(
            cur, im2col, H, W, H, W, k, p, Kreal, Ntiles);

        long long tiles = (long long)(Mtiles / 4) * Ntiles;
        gemm_bin_wmma<<<(unsigned)((tiles + 7) / 8), 256, 0, stream>>>(
            wpack, im2col, P(li, 1), convb, Mtiles, Ntiles, Ktiles, N);

        int OH = pool ? H / 2 : H, OW = pool ? W / 2 : W;
        long long tot = (long long)cout * 64 * OH * OW;
        pool_bn_clip_kernel<<<(unsigned)((tot + 255) / 256), 256, 0, stream>>>(
            convb, nxt, P(li, 2), P(li, 3), P(li, 4), P(li, 5), cout, H, W, pool);

        H = OH; W = OW;
        float* tswap = cur; cur = nxt; nxt = tswap;
    }

    head_kernel<<<(64 * 1000 + 255) / 256, 256, 0, stream>>>(
        cur, head_w, head_b, (float*)d_out);
}
